// FeatureNet_44263932952865
// MI455X (gfx1250) — compile-verified, matrix operand order checked by reading, not run on hardware
//
#include <hip/hip_runtime.h>
#include <hip/hip_bf16.h>
#include <math.h>

// ---------------------------------------------------------------------------
// Types for CDNA5 WMMA (wave32): D(f32 16x16) = A(bf16 16x32) * B(bf16 32x16) + C
// ---------------------------------------------------------------------------
typedef __attribute__((ext_vector_type(16))) __bf16 v16bf;
typedef __attribute__((ext_vector_type(8)))  float  v8f;
typedef __attribute__((ext_vector_type(4)))  unsigned int u32x4;
typedef __attribute__((ext_vector_type(8)))  unsigned int u32x8;

union BF16Frag { v16bf v; unsigned short u[16]; uint4 q[2]; };

__device__ __forceinline__ unsigned short f2bf(float f) {
  union { float f; unsigned u; } x; x.f = f;
  unsigned r = x.u + 0x7FFFu + ((x.u >> 16) & 1u);   // round-to-nearest-even
  return (unsigned short)(r >> 16);
}
__device__ __forceinline__ float bf2f(unsigned short h) {
  union { unsigned u; float f; } x; x.u = ((unsigned)h) << 16;
  return x.f;
}

// ---------------------------------------------------------------------------
// Tensor Data Mover: flat 1-D copy of `elems` bf16 values global -> LDS.
// D# group0/group1 built in SGPRs (wave-uniform); 2-group form (<=2D tensor).
// ---------------------------------------------------------------------------
__device__ __forceinline__ void tdm_copy_to_lds(const void* gsrc, void* lds_dst,
                                                unsigned elems /* < 65536 */) {
  unsigned lds_addr = (unsigned)(size_t)lds_dst;   // low 32 bits = LDS byte offset
  unsigned long long ga = (unsigned long long)(size_t)gsrc;
  u32x4 g0;
  g0[0] = 1u;                                           // count=1, user mode
  g0[1] = lds_addr;                                     // lds_addr [63:32]
  g0[2] = (unsigned)(ga & 0xFFFFFFFFu);                 // global_addr lo
  g0[3] = (unsigned)((ga >> 32) & 0x01FFFFFFu) | (2u << 30);  // addr hi | type=2
  u32x8 g1;
  g1[0] = 0x00010000u;                                  // wg_mask=0, data_size=1 (2B)
  g1[1] = (elems & 0xFFFFu) << 16;                      // tensor_dim0 lo16
  g1[2] = ((elems >> 16) & 0xFFFFu) | (1u << 16);       // tensor_dim0 hi | tensor_dim1=1
  g1[3] = (elems & 0xFFFFu) << 16;                      // tensor_dim1 hi=0 | tile_dim0
  g1[4] = 1u;                                           // tile_dim1=1, tile_dim2=0
  g1[5] = elems;                                        // tensor_dim0_stride lo32
  g1[6] = 0u;
  g1[7] = 0u;
  asm volatile("tensor_load_to_lds %0, %1" :: "s"(g0), "s"(g1) : "memory");
}

// ---------------------------------------------------------------------------
// Weight pre-pack: OIHW f32 -> bf16 in WMMA B-fragment order:
//   [ocblk][kchunk][lane(32)][elem(16)], K index = (ky*KW+kx)*C + ci.
// A lane's 16 elems are contiguous (32 bytes) -> two ds_load_b128 in the GEMM.
// ---------------------------------------------------------------------------
__global__ void pack_w_frag_k(const float* __restrict__ w, unsigned short* __restrict__ o,
                              int OC, int C, int KH, int KW,
                              int Ktrue, int Kpad, int OCpad) {
  int idx = blockIdx.x * blockDim.x + threadIdx.x;
  int nkc = Kpad >> 5;
  int ocblks = OCpad >> 4;
  int total = ocblks * nkc * 512;
  if (idx >= total) return;
  int e    = idx & 15;
  int lane = (idx >> 4) & 31;
  int cc   = (idx >> 9) % nkc;
  int ob   = (idx >> 9) / nkc;
  int hi   = lane >> 4;
  int koffv = (e < 8 ? e : e + 8) + (hi ? 8 : 0);
  int kk = (cc << 5) + koffv;
  int co = (ob << 4) + (lane & 15);
  unsigned short v = 0;
  if (kk < Ktrue && co < OC) {
    int tap = kk / C;
    int ci  = kk - tap * C;
    int ky  = tap / KW;
    int kx  = tap - ky * KW;
    v = f2bf(w[(((long long)co * C + ci) * KH + ky) * KW + kx]);
  }
  o[idx] = v;
}

// ---------------------------------------------------------------------------
// Layout shuffles
// ---------------------------------------------------------------------------
__global__ void nchw_to_nhwc_bf16_k(const float* __restrict__ in, unsigned short* __restrict__ o,
                                    int Nn, int C, int H, int W, long long total) {
  long long idx = (long long)blockIdx.x * blockDim.x + threadIdx.x;
  if (idx >= total) return;
  long long hw = (long long)H * W, chw = (long long)C * hw;
  int n = (int)(idx / chw);
  long long r = idx - (long long)n * chw;
  int c = (int)(r / hw);
  long long r2 = r - (long long)c * hw;
  int y = (int)(r2 / W), x = (int)(r2 - (long long)(r2 / W) * W);
  o[(((long long)n * H + y) * W + x) * C + c] = f2bf(in[idx]);
}

__global__ void nhwc_to_nchw_k(const float* __restrict__ in, float* __restrict__ o,
                               int Nn, int C, int H, int W, long long total) {
  long long idx = (long long)blockIdx.x * blockDim.x + threadIdx.x;
  if (idx >= total) return;
  long long p = idx / C;
  int c = (int)(idx - p * C);
  long long hw = (long long)H * W;
  int n = (int)(p / hw);
  long long r = p - (long long)n * hw;
  int y = (int)(r / W), x = (int)(r - (long long)(r / W) * W);
  o[(((long long)n * C + c) * H + y) * W + x] = in[idx];
}

__global__ void zero_f32_k(float* __restrict__ p, int n) {
  int i = blockIdx.x * blockDim.x + threadIdx.x;
  if (i < n) p[i] = 0.f;
}

// ---------------------------------------------------------------------------
// BatchNorm (biased var over N,H,W) two-pass: stats via LDS + global f32 atomics
// ---------------------------------------------------------------------------
__global__ void bn_stats_k(const float* __restrict__ in, float* __restrict__ st,
                           long long total, int C) {
  __shared__ float ls[32];
  __shared__ float lq[32];
  int tid = threadIdx.x;
  if (tid < C) { ls[tid] = 0.f; lq[tid] = 0.f; }
  __syncthreads();
  long long stride = (long long)gridDim.x * blockDim.x;
  for (long long idx = (long long)blockIdx.x * blockDim.x + tid; idx < total; idx += stride) {
    long long pp = idx / C;
    int c = (int)(idx - pp * C);
    float v = in[idx];
    atomicAdd(&ls[c], v);
    atomicAdd(&lq[c], v * v);
  }
  __syncthreads();
  if (tid < C) { atomicAdd(&st[tid], ls[tid]); atomicAdd(&st[C + tid], lq[tid]); }
}

__global__ void bn_apply_k(const float* __restrict__ in, const float* __restrict__ st,
                           unsigned short* __restrict__ ob, float* __restrict__ of,
                           long long total, int C, float Pf) {
  long long idx = (long long)blockIdx.x * blockDim.x + threadIdx.x;
  if (idx >= total) return;
  long long pp = idx / C;
  int c = (int)(idx - pp * C);
  float mean = st[c] / Pf;
  float var  = st[C + c] / Pf - mean * mean;
  float sc   = rsqrtf(var + 1e-5f);
  float y    = (in[idx] - mean) * sc;
  y = fmaxf(y, 0.f);
  ob[idx] = f2bf(y);
  if (of) of[idx] = y;
}

// ---------------------------------------------------------------------------
// up2(coarse) + fine ; write f32 (optional) and bf16
// ---------------------------------------------------------------------------
__global__ void up2_add_k(const float* __restrict__ a, const float* __restrict__ b,
                          float* __restrict__ of, unsigned short* __restrict__ ob,
                          int Nn, int Hf, int Wf, int C, long long total) {
  long long idx = (long long)blockIdx.x * blockDim.x + threadIdx.x;
  if (idx >= total) return;
  long long p = idx / C;
  int c = (int)(idx - p * C);
  long long hw = (long long)Hf * Wf;
  int n = (int)(p / hw);
  long long r = p - (long long)n * hw;
  int y = (int)(r / Wf), x = (int)(r - (long long)(r / Wf) * Wf);
  long long pl = (((long long)n * (Hf >> 1)) + (y >> 1)) * (Wf >> 1) + (x >> 1);
  float v = a[idx] + b[pl * C + c];
  if (of) of[idx] = v;
  ob[idx] = f2bf(v);
}

// ---------------------------------------------------------------------------
// Stem conv (C=3 only): scalar gather path, one 16-OC block.
// ---------------------------------------------------------------------------
__global__ void conv_stem_k(const unsigned short* __restrict__ in,
                            const unsigned short* __restrict__ wfrag,
                            float* __restrict__ out,
                            int Nn, int Hin, int Win, int C,
                            int Hout, int Wout, int OC, int OCpad,
                            int KW, int stride, int pad,
                            int Ktrue, int Kpad, int mtiles, long long Ptot) {
  __shared__ __align__(16) unsigned short shw[512];     // Kpad=32, OCpad=16
  unsigned totalElems = (unsigned)(Kpad * OCpad);
  if ((threadIdx.x >> 5) == 0) {
    tdm_copy_to_lds(wfrag, (void*)&shw[0], totalElems);
    __builtin_amdgcn_s_wait_tensorcnt(0);
  }
  __syncthreads();

  int wave = blockIdx.x * (blockDim.x >> 5) + (threadIdx.x >> 5);
  if (wave >= mtiles) return;
  int lane = threadIdx.x & 31;
  int m  = lane & 15;
  int hi = lane >> 4;

  long long p = (long long)wave * 16 + m;
  bool pv = p < Ptot;
  long long hw = (long long)Hout * Wout;
  long long pc = pv ? p : 0;
  int n = (int)(pc / hw);
  long long rr = pc - (long long)n * hw;
  int oy = (int)(rr / Wout);
  int ox = (int)(rr - (long long)oy * Wout);

  v8f acc = {0.f, 0.f, 0.f, 0.f, 0.f, 0.f, 0.f, 0.f};
  BF16Frag a, b;
  {
    const uint4* bp = (const uint4*)(shw + (lane << 4));
    b.q[0] = bp[0];
    b.q[1] = bp[1];
  }
#pragma unroll
  for (int i = 0; i < 16; ++i) {
    int kabs = (i < 8 ? i : i + 8) + (hi ? 8 : 0);
    unsigned short v = 0;
    if (pv && kabs < Ktrue) {
      int tap = kabs / C;
      int ci  = kabs - tap * C;
      int ky  = tap / KW;
      int kx  = tap - ky * KW;
      int iy = oy * stride - pad + ky;
      int ix = ox * stride - pad + kx;
      if (iy >= 0 && iy < Hin && ix >= 0 && ix < Win)
        v = in[(((long long)n * Hin + iy) * Win + ix) * C + ci];
    }
    a.u[i] = v;
  }
  acc = __builtin_amdgcn_wmma_f32_16x16x32_bf16(false, a.v, false, b.v,
                                                (short)0, acc, false, false);
#pragma unroll
  for (int r = 0; r < 8; ++r) {
    int M = r + (hi ? 8 : 0);
    long long pp = (long long)wave * 16 + M;
    if (pp < Ptot && m < OC)
      out[pp * OC + m] = acc[r];
  }
}

// ---------------------------------------------------------------------------
// Implicit-GEMM conv via v_wmma_f32_16x16x32_bf16 (C % 8 == 0).
//  - Weights TDM-staged to LDS in fragment order.
//  - One A gather (two conditional global_load_b128) feeds ALL OC blocks:
//    up to two WMMA tiles per K-chunk -> half the A traffic for OC=32 layers.
// ---------------------------------------------------------------------------
__global__ void conv_wmma_k(const unsigned short* __restrict__ in,
                            const unsigned short* __restrict__ wfrag,
                            const float* __restrict__ bias,
                            float* __restrict__ out,
                            int Nn, int Hin, int Win, int C,
                            int Hout, int Wout, int OC, int OCpad,
                            int KW, int stride, int pad,
                            int Ktrue, int Kpad, int mtiles, long long Ptot) {
  __shared__ __align__(16) unsigned short shw[13312];   // up to 26.6 KB (Kpad*OCpad)
  int nkc = Kpad >> 5;
  int ocblks = OCpad >> 4;                              // 1 or 2
  unsigned totalElems = (unsigned)(Kpad * OCpad);

  if ((threadIdx.x >> 5) == 0) {                        // wave 0 drives the TDM
    tdm_copy_to_lds(wfrag, (void*)&shw[0], totalElems);
    __builtin_amdgcn_s_wait_tensorcnt(0);
  }
  __syncthreads();

  int wave = blockIdx.x * (blockDim.x >> 5) + (threadIdx.x >> 5);
  if (wave >= mtiles) return;
  int lane = threadIdx.x & 31;
  int m  = lane & 15;
  int hi = lane >> 4;

  long long p = (long long)wave * 16 + m;
  bool pv = p < Ptot;
  long long hw = (long long)Hout * Wout;
  long long pc = pv ? p : 0;
  int n = (int)(pc / hw);
  long long rr = pc - (long long)n * hw;
  int oy = (int)(rr / Wout);
  int ox = (int)(rr - (long long)oy * Wout);

  v8f acc0 = {0.f, 0.f, 0.f, 0.f, 0.f, 0.f, 0.f, 0.f};
  v8f acc1 = {0.f, 0.f, 0.f, 0.f, 0.f, 0.f, 0.f, 0.f};
  bool two = ocblks > 1;

  for (int cc = 0; cc < nkc; ++cc) {
    BF16Frag a;
    int kc = cc << 5;
#pragma unroll
    for (int g = 0; g < 2; ++g) {
      int kabs0 = kc + (hi ? 8 : 0) + (g ? 16 : 0);
      uint4 av = {0u, 0u, 0u, 0u};
      if (pv && kabs0 < Ktrue) {
        int tap = kabs0 / C;
        int ci0 = kabs0 - tap * C;
        int ky = tap / KW, kx = tap - ky * KW;
        int iy = oy * stride - pad + ky;
        int ix = ox * stride - pad + kx;
        if (iy >= 0 && iy < Hin && ix >= 0 && ix < Win)
          av = *(const uint4*)(in + (((long long)n * Hin + iy) * Win + ix) * C + ci0);
      }
      a.q[g] = av;
    }
    BF16Frag b0;
    {
      const uint4* bp = (const uint4*)(shw + (((cc << 5) + lane) << 4));
      b0.q[0] = bp[0];
      b0.q[1] = bp[1];
    }
    acc0 = __builtin_amdgcn_wmma_f32_16x16x32_bf16(false, a.v, false, b0.v,
                                                   (short)0, acc0, false, false);
    if (two) {
      BF16Frag b1;
      const uint4* bp = (const uint4*)(shw + ((((nkc + cc) << 5) + lane) << 4));
      b1.q[0] = bp[0];
      b1.q[1] = bp[1];
      acc1 = __builtin_amdgcn_wmma_f32_16x16x32_bf16(false, a.v, false, b1.v,
                                                     (short)0, acc1, false, false);
    }
  }

#pragma unroll
  for (int r = 0; r < 8; ++r) {
    int M = r + (hi ? 8 : 0);
    long long pp = (long long)wave * 16 + M;
    if (pp < Ptot) {
      if (m < OC)
        out[pp * OC + m] = acc0[r] + (bias ? bias[m] : 0.f);
      if (two && (16 + m) < OC)
        out[pp * OC + 16 + m] = acc1[r] + (bias ? bias[16 + m] : 0.f);
    }
  }
}

// ---------------------------------------------------------------------------
// Deformable conv (C fixed = 32): per wave 16 pixels x ALL OC blocks; per tap
// build the A fragment once from masked bilinear gathers (4 corners x 2 groups
// of 8 contiguous channels -> 8 conditional global_load_b128), B from the
// TDM-staged LDS fragments, one or two v_wmma per tap.
// ---------------------------------------------------------------------------
__global__ void dcn_wmma_k(const unsigned short* __restrict__ feat,
                           const float* __restrict__ om,           // NHWC f32, 27 ch
                           const unsigned short* __restrict__ wfrag,
                           const float* __restrict__ bias,
                           float* __restrict__ out,
                           int Nn, int H, int W, int OC, int OCpad,
                           int mtiles, long long Ptot) {
  __shared__ __align__(16) unsigned short shd[9216];    // 288 * OCpad(<=32)
  const int nkc = 9;
  int ocblks = OCpad >> 4;
  unsigned totalElems = (unsigned)(288 * OCpad);

  if ((threadIdx.x >> 5) == 0) {
    tdm_copy_to_lds(wfrag, (void*)&shd[0], totalElems);
    __builtin_amdgcn_s_wait_tensorcnt(0);
  }
  __syncthreads();

  int wave = blockIdx.x * (blockDim.x >> 5) + (threadIdx.x >> 5);
  if (wave >= mtiles) return;
  int lane = threadIdx.x & 31;
  int m  = lane & 15;
  int hi = lane >> 4;
  const int C = 32;

  long long p = (long long)wave * 16 + m;
  bool pv = p < Ptot;
  long long hw = (long long)H * W;
  long long pc = pv ? p : 0;
  int n = (int)(pc / hw);
  long long rr = pc - (long long)n * hw;
  int oy = (int)(rr / W);
  int ox = (int)(rr - (long long)oy * W);

  v8f acc0 = {0.f, 0.f, 0.f, 0.f, 0.f, 0.f, 0.f, 0.f};
  v8f acc1 = {0.f, 0.f, 0.f, 0.f, 0.f, 0.f, 0.f, 0.f};
  bool two = ocblks > 1;

  for (int k = 0; k < 9; ++k) {
    int ky = k / 3, kx = k - ky * 3;
    float dyv = 0.f, dxv = 0.f, mk = 0.f;
    if (pv) {
      const float* omp = om + p * 27;
      dyv = omp[k];
      dxv = omp[9 + k];
      mk  = 1.f / (1.f + __expf(-omp[18 + k]));
    }
    float yy = (float)(oy + ky - 1) + dyv;
    float xx = (float)(ox + kx - 1) + dxv;
    float y0f = floorf(yy), x0f = floorf(xx);
    float ty = yy - y0f, tx = xx - x0f;
    int y0 = (int)y0f, x0 = (int)x0f;
    float w00 = (1.f - ty) * (1.f - tx) * mk;
    float w01 = (1.f - ty) * tx * mk;
    float w10 = ty * (1.f - tx) * mk;
    float w11 = ty * tx * mk;

    long long b00 = 0, b01 = 0, b10 = 0, b11 = 0;
    bool v00 = false, v01 = false, v10 = false, v11 = false;
    if (pv) {
      auto mkbase = [&](int yi, int xi, bool& vld) -> long long {
        vld = (yi >= 0) && (yi < H) && (xi >= 0) && (xi < W);
        int yc = yi < 0 ? 0 : (yi >= H ? H - 1 : yi);
        int xc = xi < 0 ? 0 : (xi >= W ? W - 1 : xi);
        return (((long long)n * H + yc) * W + xc) * C;
      };
      b00 = mkbase(y0,     x0,     v00);
      b01 = mkbase(y0,     x0 + 1, v01);
      b10 = mkbase(y0 + 1, x0,     v10);
      b11 = mkbase(y0 + 1, x0 + 1, v11);
    }

    BF16Frag a;
#pragma unroll
    for (int g = 0; g < 2; ++g) {
      int ci0 = (hi ? 8 : 0) + (g ? 16 : 0);
      union { uint4 q; unsigned short u[8]; } q00, q01, q10, q11;
      const uint4 z4 = {0u, 0u, 0u, 0u};
      q00.q = v00 ? *(const uint4*)(feat + b00 + ci0) : z4;
      q01.q = v01 ? *(const uint4*)(feat + b01 + ci0) : z4;
      q10.q = v10 ? *(const uint4*)(feat + b10 + ci0) : z4;
      q11.q = v11 ? *(const uint4*)(feat + b11 + ci0) : z4;
#pragma unroll
      for (int e = 0; e < 8; ++e) {
        float s = w00 * bf2f(q00.u[e]) + w01 * bf2f(q01.u[e]) +
                  w10 * bf2f(q10.u[e]) + w11 * bf2f(q11.u[e]);
        a.u[g * 8 + e] = f2bf(s);
      }
    }

    BF16Frag b0;
    {
      const uint4* bp = (const uint4*)(shd + (((k << 5) + lane) << 4));
      b0.q[0] = bp[0];
      b0.q[1] = bp[1];
    }
    acc0 = __builtin_amdgcn_wmma_f32_16x16x32_bf16(false, a.v, false, b0.v,
                                                   (short)0, acc0, false, false);
    if (two) {
      BF16Frag b1;
      const uint4* bp = (const uint4*)(shd + ((((nkc + k) << 5) + lane) << 4));
      b1.q[0] = bp[0];
      b1.q[1] = bp[1];
      acc1 = __builtin_amdgcn_wmma_f32_16x16x32_bf16(false, a.v, false, b1.v,
                                                     (short)0, acc1, false, false);
    }
  }

#pragma unroll
  for (int r = 0; r < 8; ++r) {
    int M = r + (hi ? 8 : 0);
    long long pp = (long long)wave * 16 + M;
    if (pp < Ptot) {
      if (m < OC)
        out[pp * OC + m] = acc0[r] + (bias ? bias[m] : 0.f);
      if (two && (16 + m) < OC)
        out[pp * OC + 16 + m] = acc1[r] + (bias ? bias[16 + m] : 0.f);
    }
  }
}

// ---------------------------------------------------------------------------
// Host orchestration
// ---------------------------------------------------------------------------
static inline int ceil32i(int x) { return (x + 31) & ~31; }
static inline int ceil16i(int x) { return (x + 15) & ~15; }

extern "C" void kernel_launch(void* const* d_in, const int* in_sizes, int n_in,
                              void* d_out, int out_size, void* d_ws, size_t ws_size,
                              hipStream_t stream) {
  (void)in_sizes; (void)n_in; (void)out_size; (void)ws_size;
  const int Nn = 2;
  const int H0 = 512, W0 = 640, H1 = 256, W1 = 320, H2 = 128, W2 = 160;
  const long long P0 = (long long)Nn * H0 * W0;
  const long long P1 = (long long)Nn * H1 * W1;
  const long long P2 = (long long)Nn * H2 * W2;

  char* ws = (char*)d_ws;
  size_t off = 0;
  auto alloc = [&](size_t bytes) -> char* {
    char* pp = ws + off;
    off = (off + bytes + 255) & ~(size_t)255;
    return pp;
  };

  unsigned short* xh   = (unsigned short*)alloc(P0 * 3 * 2);
  unsigned short* c0b  = (unsigned short*)alloc(P0 * 8 * 2);
  unsigned short* c1b  = (unsigned short*)alloc(P1 * 16 * 2);
  unsigned short* c2b  = (unsigned short*)alloc(P2 * 32 * 2);
  float*          c2f  = (float*)alloc(P2 * 32 * 4);
  float*          in1f = (float*)alloc(P1 * 32 * 4);
  float*          bufA = (float*)alloc(P0 * 32 * 4);           // conv/dcn f32 scratch
  unsigned short* bufB = (unsigned short*)alloc(P0 * 32 * 2);  // bf16 activation scratch
  float*          bufC = (float*)alloc(P0 * 27 * 4);           // dcn offset maps
  float*          stats = (float*)alloc(64 * 4);

  struct PK { unsigned short* p; int Kpad, OCpad, Ktrue, OC, C, KW; };
  auto pack = [&](const float* w, int OC, int C, int KH, int KW) -> PK {
    PK pk;
    pk.OC = OC; pk.C = C; pk.KW = KW;
    pk.Ktrue = KH * KW * C;
    pk.Kpad = ceil32i(pk.Ktrue);
    pk.OCpad = ceil16i(OC);
    pk.p = (unsigned short*)alloc((size_t)pk.Kpad * pk.OCpad * 2);
    int total = pk.Kpad * pk.OCpad;
    pack_w_frag_k<<<(total + 255) / 256, 256, 0, stream>>>(w, pk.p, OC, C, KH, KW,
                                                           pk.Ktrue, pk.Kpad, pk.OCpad);
    return pk;
  };

  auto conv = [&](const unsigned short* in, const PK& pk, const float* bias, float* outp,
                  int Hin, int Win, int Hout, int Wout, int stride, int pad) {
    long long P = (long long)Nn * Hout * Wout;
    int mtiles = (int)((P + 15) / 16);
    int blocks = (mtiles + 7) / 8;
    conv_wmma_k<<<blocks, 256, 0, stream>>>(in, pk.p, bias, outp,
                                            Nn, Hin, Win, pk.C, Hout, Wout,
                                            pk.OC, pk.OCpad, pk.KW, stride, pad,
                                            pk.Ktrue, pk.Kpad, mtiles, P);
  };

  auto bnrelu = [&](const float* inp, unsigned short* outb, float* outf, long long P, int C) {
    zero_f32_k<<<1, 64, 0, stream>>>(stats, 64);
    long long total = P * C;
    bn_stats_k<<<4096, 256, 0, stream>>>(inp, stats, total, C);
    int blocks = (int)((total + 255) / 256);
    bn_apply_k<<<blocks, 256, 0, stream>>>(inp, stats, outb, outf, total, C, (float)P);
  };

  auto dcn = [&](const unsigned short* feat, const PK& pkw, const PK& pkow,
                 const float* bias, const float* obias, float* outp, int H, int W) {
    conv(feat, pkow, obias, bufC, H, W, H, W, 1, 1);    // offset/mask conv -> 27 ch
    long long P = (long long)Nn * H * W;
    int mtiles = (int)((P + 15) / 16);
    int blocks = (mtiles + 7) / 8;
    dcn_wmma_k<<<blocks, 256, 0, stream>>>(feat, bufC, pkw.p, bias, outp,
                                           Nn, H, W, pkw.OC, pkw.OCpad, mtiles, P);
  };

  // ---- weight packing ----
  PK pk_c00 = pack((const float*)d_in[1],  8,  3, 3, 3);
  PK pk_c01 = pack((const float*)d_in[2],  8,  8, 3, 3);
  PK pk_c10 = pack((const float*)d_in[3], 16,  8, 5, 5);
  PK pk_c11 = pack((const float*)d_in[4], 16, 16, 3, 3);
  PK pk_c12 = pack((const float*)d_in[5], 16, 16, 3, 3);
  PK pk_c20 = pack((const float*)d_in[6], 32, 16, 5, 5);
  PK pk_c21 = pack((const float*)d_in[7], 32, 32, 3, 3);
  PK pk_c22 = pack((const float*)d_in[8], 32, 32, 3, 3);
  PK pk_o1  = pack((const float*)d_in[9],  32, 32, 1, 1);
  PK pk_o2  = pack((const float*)d_in[10], 32, 32, 3, 3);
  PK pk_o3  = pack((const float*)d_in[11], 32, 32, 3, 3);
  PK pk_i1  = pack((const float*)d_in[12], 32, 16, 1, 1);
  const float* i1b = (const float*)d_in[13];
  PK pk_i2  = pack((const float*)d_in[14], 32,  8, 1, 1);
  const float* i2b = (const float*)d_in[15];

  const int dcn_oc[9] = {32, 32, 32, 32, 32, 16, 32, 32, 8};
  PK pk_dw[9], pk_dow[9];
  const float *db[9], *dob[9];
  for (int j = 0; j < 9; ++j) {
    pk_dw[j]  = pack((const float*)d_in[16 + 4 * j + 0], dcn_oc[j], 32, 3, 3);
    db[j]     = (const float*)d_in[16 + 4 * j + 1];
    pk_dow[j] = pack((const float*)d_in[16 + 4 * j + 2], 27, 32, 3, 3);
    dob[j]    = (const float*)d_in[16 + 4 * j + 3];
  }

  // ---- input NCHW -> NHWC bf16 ----
  {
    long long t = P0 * 3;
    nchw_to_nhwc_bf16_k<<<(int)((t + 255) / 256), 256, 0, stream>>>(
        (const float*)d_in[0], xh, Nn, 3, H0, W0, t);
  }

  // ---- backbone ----
  {   // stem: C=3 scalar-gather kernel
    int mtiles = (int)((P0 + 15) / 16);
    conv_stem_k<<<(mtiles + 7) / 8, 256, 0, stream>>>(
        xh, pk_c00.p, bufA, Nn, H0, W0, 3, H0, W0, 8, 16, 3, 1, 1,
        pk_c00.Ktrue, pk_c00.Kpad, mtiles, P0);
  }
  bnrelu(bufA, bufB, nullptr, P0, 8);
  conv(bufB, pk_c01, nullptr, bufA, H0, W0, H0, W0, 1, 1); bnrelu(bufA, c0b,  nullptr, P0, 8);

  conv(c0b,  pk_c10, nullptr, bufA, H0, W0, H1, W1, 2, 2); bnrelu(bufA, bufB, nullptr, P1, 16);
  conv(bufB, pk_c11, nullptr, bufA, H1, W1, H1, W1, 1, 1); bnrelu(bufA, bufB, nullptr, P1, 16);
  conv(bufB, pk_c12, nullptr, bufA, H1, W1, H1, W1, 1, 1); bnrelu(bufA, c1b,  nullptr, P1, 16);

  conv(c1b,  pk_c20, nullptr, bufA, H1, W1, H2, W2, 2, 2); bnrelu(bufA, bufB, nullptr, P2, 32);
  conv(bufB, pk_c21, nullptr, bufA, H2, W2, H2, W2, 1, 1); bnrelu(bufA, bufB, nullptr, P2, 32);
  conv(bufB, pk_c22, nullptr, bufA, H2, W2, H2, W2, 1, 1); bnrelu(bufA, c2b,  c2f,     P2, 32);

  // ---- head 1 (1/4 res) -> feat2 ----
  conv(c2b, pk_o1, nullptr, bufA, H2, W2, H2, W2, 1, 0);   bnrelu(bufA, bufB, nullptr, P2, 32);
  dcn(bufB, pk_dw[0], pk_dow[0], db[0], dob[0], bufA, H2, W2); bnrelu(bufA, bufB, nullptr, P2, 32);
  dcn(bufB, pk_dw[1], pk_dow[1], db[1], dob[1], bufA, H2, W2); bnrelu(bufA, bufB, nullptr, P2, 32);
  dcn(bufB, pk_dw[2], pk_dow[2], db[2], dob[2], bufA, H2, W2);
  {
    long long t = P2 * 32;
    nhwc_to_nchw_k<<<(int)((t + 255) / 256), 256, 0, stream>>>(
        bufA, (float*)d_out, Nn, 32, H2, W2, t);
  }

  // ---- head 2 (1/2 res) -> feat1 ----
  conv(c1b, pk_i1, i1b, bufA, H1, W1, H1, W1, 1, 0);
  {
    long long t = P1 * 32;
    up2_add_k<<<(int)((t + 255) / 256), 256, 0, stream>>>(
        bufA, c2f, in1f, bufB, Nn, H1, W1, 32, t);
  }
  conv(bufB, pk_o2, nullptr, bufA, H1, W1, H1, W1, 1, 1);  bnrelu(bufA, bufB, nullptr, P1, 32);
  dcn(bufB, pk_dw[3], pk_dow[3], db[3], dob[3], bufA, H1, W1); bnrelu(bufA, bufB, nullptr, P1, 32);
  dcn(bufB, pk_dw[4], pk_dow[4], db[4], dob[4], bufA, H1, W1); bnrelu(bufA, bufB, nullptr, P1, 32);
  dcn(bufB, pk_dw[5], pk_dow[5], db[5], dob[5], bufA, H1, W1);  // OC=16
  {
    long long t = P1 * 16;
    nhwc_to_nchw_k<<<(int)((t + 255) / 256), 256, 0, stream>>>(
        bufA, (float*)d_out + P2 * 32, Nn, 16, H1, W1, t);
  }

  // ---- head 3 (full res) -> feat0 ----
  conv(c0b, pk_i2, i2b, bufA, H0, W0, H0, W0, 1, 0);
  {
    long long t = P0 * 32;
    up2_add_k<<<(int)((t + 255) / 256), 256, 0, stream>>>(
        bufA, in1f, nullptr, bufB, Nn, H0, W0, 32, t);
  }
  conv(bufB, pk_o3, nullptr, bufA, H0, W0, H0, W0, 1, 1);  bnrelu(bufA, bufB, nullptr, P0, 32);
  dcn(bufB, pk_dw[6], pk_dow[6], db[6], dob[6], bufA, H0, W0); bnrelu(bufA, bufB, nullptr, P0, 32);
  dcn(bufB, pk_dw[7], pk_dow[7], db[7], dob[7], bufA, H0, W0); bnrelu(bufA, bufB, nullptr, P0, 32);
  dcn(bufB, pk_dw[8], pk_dow[8], db[8], dob[8], bufA, H0, W0);  // OC=8
  {
    long long t = P0 * 8;
    nhwc_to_nchw_k<<<(int)((t + 255) / 256), 256, 0, stream>>>(
        bufA, (float*)d_out + P2 * 32 + P1 * 16, Nn, 8, H0, W0, t);
  }
}
